// RefineBlock_62062277427560
// MI455X (gfx1250) — matrix operand order repaired and checked by reading, not compiled, not run on hardware
//
#include <hip/hip_runtime.h>
#include <hip/hip_bf16.h>
#include <math.h>

typedef _Float16 v16h __attribute__((ext_vector_type(16)));
typedef _Float16 h8   __attribute__((ext_vector_type(8)));
typedef float    v8f  __attribute__((ext_vector_type(8)));

#define BB   4
#define NN   4096
#define CC   64
#define KK1  4
#define KK2  6
#define EPSV 1e-5f

// ---------------------------------------------------------------- helpers
__device__ inline void qmul(const float a[4], const float b[4], float r[4]) {
    r[0] = a[0]*b[0] - a[1]*b[1] - a[2]*b[2] - a[3]*b[3];
    r[1] = a[0]*b[1] + a[1]*b[0] + a[2]*b[3] - a[3]*b[2];
    r[2] = a[0]*b[2] - a[1]*b[3] + a[2]*b[0] + a[3]*b[1];
    r[3] = a[0]*b[3] + a[1]*b[2] - a[2]*b[1] + a[3]*b[0];
}
__device__ inline void qrot(const float q[4], const float v[3], float out[3]) {
    float vq[4] = {0.f, v[0], v[1], v[2]};
    float n2 = q[0]*q[0] + q[1]*q[1] + q[2]*q[2] + q[3]*q[3] + 1e-10f;
    float iq[4] = {q[0]/n2, -q[1]/n2, -q[2]/n2, -q[3]/n2};
    float t1[4], r[4];
    qmul(q, vq, t1);
    qmul(t1, iq, r);
    out[0] = r[1]; out[1] = r[2]; out[2] = r[3];
}

// ---------------------------------------------------------------- transform
__global__ void k_transform(const float* __restrict__ pos1,
                            const float* __restrict__ qc,
                            const float* __restrict__ tc,
                            float* __restrict__ pos1w) {
    int i = blockIdx.x * blockDim.x + threadIdx.x;
    if (i >= BB * NN) return;
    int b = i / NN;
    float q[4] = {qc[b*4], qc[b*4+1], qc[b*4+2], qc[b*4+3]};
    float v[3] = {pos1[i*3], pos1[i*3+1], pos1[i*3+2]};
    float o[3];
    qrot(q, v, o);
    pos1w[i*3+0] = o[0] + tc[b*3+0];
    pos1w[i*3+1] = o[1] + tc[b*3+1];
    pos1w[i*3+2] = o[2] + tc[b*3+2];
}

// ---------------------------------------------------------------- brute KNN
// whole database batch (4096 pts * 12B = 48KB) lives in LDS (320KB/WGP)
template<int KNB>
__global__ __launch_bounds__(256) void k_knn(const float* __restrict__ qpts,
                                             const float* __restrict__ db,
                                             int* __restrict__ outIdx) {
    __shared__ float sx[NN], sy[NN], sz[NN];
    int b = blockIdx.y;
    const float* dbb = db + (size_t)b * NN * 3;
    for (int i = threadIdx.x; i < NN; i += blockDim.x) {
        sx[i] = dbb[i*3+0]; sy[i] = dbb[i*3+1]; sz[i] = dbb[i*3+2];
    }
    __syncthreads();
    int qn = blockIdx.x * blockDim.x + threadIdx.x;
    const float* qp = qpts + ((size_t)b * NN + qn) * 3;
    float qx = qp[0], qy = qp[1], qz = qp[2];
    float bd[KNB]; int bi[KNB];
#pragma unroll
    for (int j = 0; j < KNB; ++j) { bd[j] = 3.4e38f; bi[j] = 0; }
    for (int m = 0; m < NN; ++m) {
        float dx = qx - sx[m], dy = qy - sy[m], dz = qz - sz[m];
        float d = dx*dx + dy*dy + dz*dz;
        if (d < bd[KNB-1]) {
            int j = KNB - 1;
            while (j > 0 && bd[j-1] > d) { bd[j] = bd[j-1]; bi[j] = bi[j-1]; --j; }
            bd[j] = d; bi[j] = m;
        }
    }
    int* op = outIdx + ((size_t)b * NN + qn) * KNB;
#pragma unroll
    for (int j = 0; j < KNB; ++j) op[j] = bi[j];
}

// ---------------------------------------------------------------- pack stage-1
__global__ void k_pack1(const float* __restrict__ pos1w,
                        const float* __restrict__ pos2,
                        const float* __restrict__ feat1,
                        const float* __restrict__ feat2,
                        const int*   __restrict__ idx2,
                        _Float16* __restrict__ x, int total) {
    int i = blockIdx.x * blockDim.x + threadIdx.x;
    if (i >= total) return;
    int c   = i % 160;
    int row = i / 160;
    int bn  = row / KK2;
    int b   = bn / NN, n = bn % NN;
    float v = 0.f;
    if (c < 131) {
        if (c < 3) {
            int id = idx2[row];
            v = pos2[((size_t)b*NN + id)*3 + c] - pos1w[((size_t)b*NN + n)*3 + c];
        } else if (c < 67) {
            int id = idx2[row];
            v = feat2[((size_t)b*NN + id)*CC + (c-3)];
        } else {
            v = feat1[((size_t)b*NN + n)*CC + (c-67)];
        }
    }
    x[i] = (_Float16)v;
}

// ---------------------------------------------------------------- pack stage-2
__global__ void k_pack2(const float* __restrict__ pos1w,
                        const float* __restrict__ fe,
                        const float* __restrict__ feat1,
                        const int*   __restrict__ idx1,
                        _Float16* __restrict__ x, int total) {
    int i = blockIdx.x * blockDim.x + threadIdx.x;
    if (i >= total) return;
    int c   = i % 160;
    int row = i / 160;
    int bn  = row / KK1;
    int b   = bn / NN, n = bn % NN;
    float v = 0.f;
    if (c < 131) {
        if (c < 3) {
            int id = idx1[row];
            v = pos1w[((size_t)b*NN + id)*3 + c] - pos1w[((size_t)b*NN + n)*3 + c];
        } else if (c < 67) {
            int id = idx1[row];
            v = fe[((size_t)b*NN + id)*CC + (c-3)];
        } else {
            v = feat1[((size_t)b*NN + n)*CC + (c-67)];
        }
    }
    x[i] = (_Float16)v;
}

// ---------------------------------------------------------------- weight pack
// w (f32, O x Cin) -> wp (f16, O x Kpad, zero padded K). <= 40KB, L2/L0 resident.
__global__ void k_wpack(const float* __restrict__ w, _Float16* __restrict__ wp,
                        int Cin, int Kpad, int total) {
    int i = blockIdx.x * blockDim.x + threadIdx.x;
    if (i >= total) return;
    int o = i / Kpad, k = i % Kpad;
    wp[i] = (k < Cin) ? (_Float16)w[(size_t)o * Cin + k] : (_Float16)0.f;
}

// ---------------------------------------------------------------- WMMA GEMM
// y[m][o] = sum_k x[m][k] * wp[o][k]  (both f16, K padded to mult of 32).
// Each wave computes a full 16 x (NT*16) strip: one A fragment feeds NT
// back-to-back v_wmma_f32_16x16x32_f16 ops per K-step.
template<int NT>
__global__ __launch_bounds__(128) void k_gemm(const _Float16* __restrict__ xbuf,
                                              const _Float16* __restrict__ wp,
                                              _Float16* __restrict__ y,
                                              int Kpad) {
    const int O = NT * 16;
    int lane = threadIdx.x & 31;
    int wave = threadIdx.x >> 5;
    int row0 = (blockIdx.x * 4 + wave) * 16;

    const int koffA = (lane < 16) ? 0 : 16;
    const int koffB = (lane < 16) ? 0 : 16;
    const _Float16* arow = xbuf + (size_t)(row0 + (lane & 15)) * Kpad;
    const _Float16* brow = wp + (size_t)(lane & 15) * Kpad;   // + t*16*Kpad per tile

    v8f acc[NT];
#pragma unroll
    for (int t = 0; t < NT; ++t)
        acc[t] = (v8f){0.f, 0.f, 0.f, 0.f, 0.f, 0.f, 0.f, 0.f};

    for (int k0 = 0; k0 < Kpad; k0 += 32) {
        // A fragment: contiguous K = k0+koffA+j (two 16B chunks), matches B
        h8 a0 = *(const h8*)(arow + k0 + koffA);
        h8 a1 = *(const h8*)(arow + k0 + koffA + 8);
        v16h af;
#pragma unroll
        for (int j = 0; j < 8; ++j) { af[j] = a0[j]; af[j+8] = a1[j]; }
#pragma unroll
        for (int t = 0; t < NT; ++t) {
            // B fragment: contiguous 32B (halves j -> K = k0+koffB+j), aligned
            v16h bf = *(const v16h*)(brow + (size_t)t * 16 * Kpad + k0 + koffB);
            acc[t] = __builtin_amdgcn_wmma_f32_16x16x32_f16(false, af, false, bf,
                                                            (short)0, acc[t],
                                                            false, false);
        }
    }
    // D layout: VGPR r -> rows row0+r (lanes 0-15) / row0+8+r (lanes 16-31)
    int mrow = row0 + ((lane >= 16) ? 8 : 0);
#pragma unroll
    for (int t = 0; t < NT; ++t) {
        int ncol = t * 16 + (lane & 15);
#pragma unroll
        for (int r = 0; r < 8; ++r)
            y[(size_t)(mrow + r) * O + ncol] = (_Float16)acc[t][r];
    }
}

// ---------------------------------------------------------------- BN stats
__global__ __launch_bounds__(256) void k_stats(const _Float16* __restrict__ y,
                                               int M, int O, float* __restrict__ stats) {
    __shared__ float s1[256], s2[256];
    int o = blockIdx.x;
    float s = 0.f, ss = 0.f;
    for (int m = threadIdx.x; m < M; m += blockDim.x) {
        float v = (float)y[(size_t)m * O + o];
        s += v; ss += v * v;
    }
    s1[threadIdx.x] = s; s2[threadIdx.x] = ss;
    __syncthreads();
    for (int st = 128; st > 0; st >>= 1) {
        if (threadIdx.x < st) {
            s1[threadIdx.x] += s1[threadIdx.x + st];
            s2[threadIdx.x] += s2[threadIdx.x + st];
        }
        __syncthreads();
    }
    if (threadIdx.x == 0) { stats[2*o] = s1[0]; stats[2*o+1] = s2[0]; }
}

__global__ void k_finalize(const float* __restrict__ stats,
                           const float* __restrict__ g,
                           const float* __restrict__ bta,
                           float invM, int O, float* __restrict__ ss) {
    int o = blockIdx.x * blockDim.x + threadIdx.x;
    if (o >= O) return;
    float mean = stats[2*o] * invM;
    float var  = stats[2*o+1] * invM - mean * mean;
    float sc   = g[o] * rsqrtf(var + EPSV);
    ss[2*o]   = sc;
    ss[2*o+1] = bta[o] - mean * sc;
}

// ---------------------------------------------------------------- norm+relu repack
__global__ void k_normrelu(const _Float16* __restrict__ y,
                           _Float16* __restrict__ x,
                           const float* __restrict__ ss, int O, int total) {
    int i = blockIdx.x * blockDim.x + threadIdx.x;
    if (i >= total) return;
    int o = i % O;
    float v = (float)y[i] * ss[2*o] + ss[2*o+1];
    x[i] = (_Float16)fmaxf(v, 0.f);
}

// ---------------------------------------------------------------- fe = sum_k relu(bn(y3))
__global__ void k_fe(const _Float16* __restrict__ y,
                     const float* __restrict__ ss,
                     float* __restrict__ fe) {
    int i = blockIdx.x * blockDim.x + threadIdx.x;   // over B*N*64
    if (i >= BB * NN * CC) return;
    int c  = i % CC;
    int bn = i / CC;
    float sc = ss[2*c], sh = ss[2*c+1];
    float acc = 0.f;
#pragma unroll
    for (int kk = 0; kk < KK2; ++kk) {
        float v = (float)y[((size_t)bn * KK2 + kk) * CC + c] * sc + sh;
        acc += fmaxf(v, 0.f);
    }
    fe[i] = acc;
}

// ---------------------------------------------------------------- pooled = sum_{n,k} relu(bn(y5))
__global__ __launch_bounds__(256) void k_pool(const _Float16* __restrict__ y,
                                              const float* __restrict__ ss,
                                              float* __restrict__ pooled) {
    __shared__ float red[256];
    int bc = blockIdx.x;           // b*64 + c
    int b = bc / CC, c = bc % CC;
    float sc = ss[2*c], sh = ss[2*c+1];
    const int ROWS = NN * KK1;     // 16384
    float s = 0.f;
    for (int r = threadIdx.x; r < ROWS; r += blockDim.x) {
        size_t m = (size_t)b * ROWS + r;
        float v = (float)y[m * CC + c] * sc + sh;
        s += fmaxf(v, 0.f);
    }
    red[threadIdx.x] = s;
    __syncthreads();
    for (int st = 128; st > 0; st >>= 1) {
        if (threadIdx.x < st) red[threadIdx.x] += red[threadIdx.x + st];
        __syncthreads();
    }
    if (threadIdx.x == 0) pooled[bc] = red[0];
}

// ---------------------------------------------------------------- head
__global__ void k_head(const float* __restrict__ pooled,
                       const float* __restrict__ qc,
                       const float* __restrict__ tc,
                       const float* __restrict__ wq, const float* __restrict__ bq,
                       const float* __restrict__ wt, const float* __restrict__ bt,
                       float* __restrict__ out) {
    int b = threadIdx.x;
    if (b >= BB) return;
    const float* p = pooled + b * CC;
    float qd[4];
    for (int j = 0; j < 4; ++j) {
        float s = bq[j];
        for (int c = 0; c < CC; ++c) s += wq[j*CC + c] * p[c];
        qd[j] = s;
    }
    float nq = sqrtf(qd[0]*qd[0]+qd[1]*qd[1]+qd[2]*qd[2]+qd[3]*qd[3]) + 1e-10f;
    for (int j = 0; j < 4; ++j) qd[j] /= nq;
    float td[3];
    for (int j = 0; j < 3; ++j) {
        float s = bt[j];
        for (int c = 0; c < CC; ++c) s += wt[j*CC + c] * p[c];
        td[j] = s;
    }
    float qnw[4], qcb[4] = {qc[b*4], qc[b*4+1], qc[b*4+2], qc[b*4+3]};
    qmul(qd, qcb, qnw);
    float nn2 = sqrtf(qnw[0]*qnw[0]+qnw[1]*qnw[1]+qnw[2]*qnw[2]+qnw[3]*qnw[3]) + 1e-10f;
    float tcv[3] = {tc[b*3], tc[b*3+1], tc[b*3+2]};
    float tr[3];
    qrot(qd, tcv, tr);
    out[b*7+0] = qnw[0]/nn2; out[b*7+1] = qnw[1]/nn2;
    out[b*7+2] = qnw[2]/nn2; out[b*7+3] = qnw[3]/nn2;
    out[b*7+4] = tr[0] + td[0];
    out[b*7+5] = tr[1] + td[1];
    out[b*7+6] = tr[2] + td[2];
}

// ---------------------------------------------------------------- launch
extern "C" void kernel_launch(void* const* d_in, const int* in_sizes, int n_in,
                              void* d_out, int out_size, void* d_ws, size_t ws_size,
                              hipStream_t stream) {
    const float* pos1   = (const float*)d_in[0];
    const float* pos2   = (const float*)d_in[1];
    const float* feat1  = (const float*)d_in[2];
    const float* feat2  = (const float*)d_in[3];
    const float* qc     = (const float*)d_in[4];
    const float* tcoar  = (const float*)d_in[5];
    const float* w1[3]  = {(const float*)d_in[6],  (const float*)d_in[9],  (const float*)d_in[12]};
    const float* g1[3]  = {(const float*)d_in[7],  (const float*)d_in[10], (const float*)d_in[13]};
    const float* b1[3]  = {(const float*)d_in[8],  (const float*)d_in[11], (const float*)d_in[14]};
    const float* w2[2]  = {(const float*)d_in[15], (const float*)d_in[18]};
    const float* g2[2]  = {(const float*)d_in[16], (const float*)d_in[19]};
    const float* b2[2]  = {(const float*)d_in[17], (const float*)d_in[20]};
    const float* wq = (const float*)d_in[21];
    const float* bq = (const float*)d_in[22];
    const float* wt = (const float*)d_in[23];
    const float* bt = (const float*)d_in[24];
    float* out = (float*)d_out;

    char* ws = (char*)d_ws;
    const size_t off_pos1w  = 0;                       // 196608
    const size_t off_idx2   = 196608;                  // 393216
    const size_t off_idx1   = 589824;                  // 262144
    const size_t off_fe     = 851968;                  // 4 MB
    const size_t off_stats  = 5046272;                 // 1 KB
    const size_t off_ss     = 5047296;                 // 1 KB
    const size_t off_pooled = 5048320;                 // 1 KB
    const size_t off_wp     = 6ull * 1024 * 1024;      // 40 KB (f16 packed weights)
    const size_t off_bufA   = 8ull * 1024 * 1024;      // 31.5 MB used
    const size_t off_bufB   = 40ull * 1024 * 1024;     // 25.2 MB used

    float*     pos1w  = (float*)(ws + off_pos1w);
    int*       idx2   = (int*)  (ws + off_idx2);
    int*       idx1   = (int*)  (ws + off_idx1);
    float*     fe     = (float*)(ws + off_fe);
    float*     stats  = (float*)(ws + off_stats);
    float*     ssb    = (float*)(ws + off_ss);
    float*     pooled = (float*)(ws + off_pooled);
    _Float16*  wpak   = (_Float16*)(ws + off_wp);
    _Float16*  bufA   = (_Float16*)(ws + off_bufA);
    _Float16*  bufB   = (_Float16*)(ws + off_bufB);

    const int M1 = BB * NN * KK2;   // 98304
    const int M2 = BB * NN * KK1;   // 65536

    // 1. warp pos1 by coarse pose
    k_transform<<<(BB*NN + 255)/256, 256, 0, stream>>>(pos1, qc, tcoar, pos1w);

    // 2. knn of pos1w into pos2 (K=6)
    k_knn<KK2><<<dim3(NN/256, BB), 256, 0, stream>>>(pos1w, pos2, idx2);

    // 3. pack stage-1 features (f16, Kpad=160)
    k_pack1<<<(M1*160 + 255)/256, 256, 0, stream>>>(pos1w, pos2, feat1, feat2, idx2,
                                                    bufA, M1*160);

    // 4. MLP1 layer 0: 131 -> 128
    k_wpack<<<(128*160 + 255)/256, 256, 0, stream>>>(w1[0], wpak, 131, 160, 128*160);
    k_gemm<8><<<M1/64, 128, 0, stream>>>(bufA, wpak, bufB, 160);
    k_stats<<<128, 256, 0, stream>>>(bufB, M1, 128, stats);
    k_finalize<<<1, 128, 0, stream>>>(stats, g1[0], b1[0], 1.f/M1, 128, ssb);
    k_normrelu<<<(M1*128 + 255)/256, 256, 0, stream>>>(bufB, bufA, ssb, 128, M1*128);

    // 5. MLP1 layer 1: 128 -> 64
    k_wpack<<<(64*128 + 255)/256, 256, 0, stream>>>(w1[1], wpak, 128, 128, 64*128);
    k_gemm<4><<<M1/64, 128, 0, stream>>>(bufA, wpak, bufB, 128);
    k_stats<<<64, 256, 0, stream>>>(bufB, M1, 64, stats);
    k_finalize<<<1, 64, 0, stream>>>(stats, g1[1], b1[1], 1.f/M1, 64, ssb);
    k_normrelu<<<(M1*64 + 255)/256, 256, 0, stream>>>(bufB, bufA, ssb, 64, M1*64);

    // 6. MLP1 layer 2: 64 -> 64, then sum over K2 -> fe
    k_wpack<<<(64*64 + 255)/256, 256, 0, stream>>>(w1[2], wpak, 64, 64, 64*64);
    k_gemm<4><<<M1/64, 128, 0, stream>>>(bufA, wpak, bufB, 64);
    k_stats<<<64, 256, 0, stream>>>(bufB, M1, 64, stats);
    k_finalize<<<1, 64, 0, stream>>>(stats, g1[2], b1[2], 1.f/M1, 64, ssb);
    k_fe<<<(BB*NN*CC + 255)/256, 256, 0, stream>>>(bufB, ssb, fe);

    // 7. self-knn of pos1w (K=4)
    k_knn<KK1><<<dim3(NN/256, BB), 256, 0, stream>>>(pos1w, pos1w, idx1);

    // 8. pack stage-2 features
    k_pack2<<<(M2*160 + 255)/256, 256, 0, stream>>>(pos1w, fe, feat1, idx1,
                                                    bufA, M2*160);

    // 9. MLP2 layer 0: 131 -> 128
    k_wpack<<<(128*160 + 255)/256, 256, 0, stream>>>(w2[0], wpak, 131, 160, 128*160);
    k_gemm<8><<<M2/64, 128, 0, stream>>>(bufA, wpak, bufB, 160);
    k_stats<<<128, 256, 0, stream>>>(bufB, M2, 128, stats);
    k_finalize<<<1, 128, 0, stream>>>(stats, g2[0], b2[0], 1.f/M2, 128, ssb);
    k_normrelu<<<(M2*128 + 255)/256, 256, 0, stream>>>(bufB, bufA, ssb, 128, M2*128);

    // 10. MLP2 layer 1: 128 -> 64
    k_wpack<<<(64*128 + 255)/256, 256, 0, stream>>>(w2[1], wpak, 128, 128, 64*128);
    k_gemm<4><<<M2/64, 128, 0, stream>>>(bufA, wpak, bufB, 128);
    k_stats<<<64, 256, 0, stream>>>(bufB, M2, 64, stats);
    k_finalize<<<1, 64, 0, stream>>>(stats, g2[1], b2[1], 1.f/M2, 64, ssb);

    // 11. global pool over (n, k)
    k_pool<<<BB*CC, 256, 0, stream>>>(bufB, ssb, pooled);

    // 12. pose head
    k_head<<<1, 32, 0, stream>>>(pooled, qc, tcoar, wq, bq, wt, bt, out);

    (void)in_sizes; (void)n_in; (void)out_size; (void)ws_size;
}
